// SAModule_34686155883118
// MI455X (gfx1250) — matrix operand order repaired and checked by reading, not compiled, not run on hardware
//
#include <hip/hip_runtime.h>
#include <hip/hip_bf16.h>

// ---------------- problem constants (match reference) ----------------
#define BB       8        // clouds
#define PP       4096     // points per cloud
#define CIN      64       // input feature dim
#define KSAMP    1024     // centroids per cloud (P * 0.25)
#define KNEI     64       // max neighbors
#define RAD2     0.04f    // R^2, R = 0.2
#define KPAD     96       // 67 padded to multiple of 32 for bf16 WMMA
#define H1DIM    128
#define H2DIM    128
#define H3DIM    256
#define NQ_TOT   (BB * KSAMP)

// pre-converted bf16 weight pool layout (elements)
#define W1OFF    0
#define W1ELEM   (KPAD * H1DIM)          // 12288 (rows 67..95 zero)
#define W2OFF    (W1OFF + W1ELEM)
#define W2ELEM   (H1DIM * H2DIM)         // 16384
#define W3OFF    (W2OFF + W2ELEM)
#define W3ELEM   (H2DIM * H3DIM)         // 32768
#define WELEM    (W3OFF + W3ELEM)        // 61440

typedef __attribute__((ext_vector_type(8)))  float  v8f;
typedef __attribute__((ext_vector_type(16))) __bf16 v16bf;
typedef int v4i_t __attribute__((vector_size(16)));
typedef __attribute__((address_space(1))) v4i_t* gv4i_p;   // global int4*
typedef __attribute__((address_space(3))) v4i_t* lv4i_p;   // LDS int4*

union FragBF {
    v16bf v;
    uint4 q[2];
};

__device__ __forceinline__ __bf16 f32_to_bf16(float f) {
    unsigned u = __builtin_bit_cast(unsigned, f);
    unsigned r = (u + 0x7FFFu + ((u >> 16) & 1u)) >> 16;
    unsigned short s = (unsigned short)r;
    return __builtin_bit_cast(__bf16, s);
}

// ---- CDNA5 async global->LDS copy (ASYNCcnt tracked), 16B per lane ----
__device__ __forceinline__ void async_copy_b128(__bf16* lds_dst, const __bf16* gsrc) {
#if __has_builtin(__builtin_amdgcn_global_load_async_to_lds_b128)
    __builtin_amdgcn_global_load_async_to_lds_b128(
        (gv4i_p)(unsigned long long)(const void*)gsrc,
        (lv4i_p)(unsigned)(unsigned long long)(void*)lds_dst,
        0, 0);
#else
    unsigned l = (unsigned)(unsigned long long)(void*)lds_dst;
    asm volatile("global_load_async_to_lds_b128 %0, %1, off"
                 :: "v"(l), "v"((unsigned long long)(const void*)gsrc)
                 : "memory");
#endif
}

template <int N>
__device__ __forceinline__ void wait_asynccnt() {
#if __has_builtin(__builtin_amdgcn_s_wait_asynccnt)
    __builtin_amdgcn_s_wait_asynccnt((unsigned short)N);
#else
    asm volatile("s_wait_asynccnt %0" :: "i"(N) : "memory");
#endif
}

// =====================================================================
// Kernel 0: one-time f32 -> bf16 conversion of x and the (padded)
// weight matrices into workspace. Removes per-block conversion VALU and
// halves L2 traffic for the 8192 MLP workgroups.
// =====================================================================
__global__ void convert_bf16_kernel(const float* __restrict__ x,
                                    const float* __restrict__ W1,
                                    const float* __restrict__ W2,
                                    const float* __restrict__ W3,
                                    __bf16* __restrict__ xbf,
                                    __bf16* __restrict__ wbf) {
    const size_t tid    = (size_t)blockIdx.x * blockDim.x + threadIdx.x;
    const size_t stride = (size_t)gridDim.x * blockDim.x;
    const size_t NX = (size_t)BB * PP * CIN;

    for (size_t i = tid; i < NX; i += stride) xbf[i] = f32_to_bf16(x[i]);

    for (size_t i = tid; i < (size_t)WELEM; i += stride) {
        float v;
        if (i < W2OFF) {                        // W1 padded [96][128]
            size_t j = i - W1OFF;
            size_t k = j >> 7, n = j & 127;
            v = (k < (CIN + 3)) ? W1[k * H1DIM + n] : 0.0f;
        } else if (i < W3OFF) {                 // W2 [128][128]
            v = W2[i - W2OFF];
        } else {                                // W3 [128][256]
            v = W3[i - W3OFF];
        }
        wbf[i] = f32_to_bf16(v);
    }
}

// =====================================================================
// Kernel 1: farthest point sampling. One 1024-thread WG per cloud.
// Distances in registers; LDS only for the pos[last] broadcast;
// argmax with first-index tie-break packed into a 64-bit key.
// =====================================================================
__global__ void fps_kernel(const float* __restrict__ pos, int* __restrict__ idx) {
    extern __shared__ char smem[];
    float* px = (float*)smem;
    float* py = px + PP;
    float* pz = py + PP;
    unsigned long long* wred = (unsigned long long*)(pz + PP);

    const int b = blockIdx.x;
    const int t = threadIdx.x;
    const float* pb = pos + (size_t)b * PP * 3;

    float cx[4], cy[4], cz[4], dreg[4];
#pragma unroll
    for (int k = 0; k < 4; ++k) {
        int i = t + k * 1024;
        float x = pb[i * 3 + 0], y = pb[i * 3 + 1], z = pb[i * 3 + 2];
        px[i] = x; py[i] = y; pz[i] = z;
        cx[k] = x; cy[k] = y; cz[k] = z;
        dreg[k] = __builtin_inff();
    }
    __syncthreads();

    int last = 0;
    for (int s = 0; s < KSAMP; ++s) {
        if (t == 0) idx[b * KSAMP + s] = last;
        float lx = px[last], ly = py[last], lz = pz[last];

        unsigned long long best = 0;
#pragma unroll
        for (int k = 0; k < 4; ++k) {
            int i = t + k * 1024;
            float dx = cx[k] - lx, dy = cy[k] - ly, dz = cz[k] - lz;
            float d = dx * dx + dy * dy + dz * dz;
            float dm = fminf(dreg[k], d);
            dreg[k] = dm;
            unsigned long long key =
                ((unsigned long long)__float_as_uint(dm) << 32) |
                (unsigned long long)(unsigned)(PP - 1 - i);
            best = best > key ? best : key;
        }
#pragma unroll
        for (int off = 16; off > 0; off >>= 1) {
            unsigned long long o = __shfl_xor(best, off, 32);
            best = best > o ? best : o;
        }
        if ((t & 31) == 0) wred[t >> 5] = best;
        __syncthreads();
        if (t < 32) {
            unsigned long long v = wred[t];
#pragma unroll
            for (int off = 16; off > 0; off >>= 1) {
                unsigned long long o = __shfl_xor(v, off, 32);
                v = v > o ? v : o;
            }
            if (t == 0) wred[0] = v;
        }
        __syncthreads();
        last = PP - 1 - (int)(wred[0] & 0xFFFFFFFFull);
        __syncthreads();
    }
}

// =====================================================================
// Kernel 2: ordered radius search, one wave32 per query.
// ballot + prefix popcount -> first <=64 in-range points in index order.
// =====================================================================
__global__ void radius_kernel(const float* __restrict__ pos,
                              const int* __restrict__ idx,
                              int* __restrict__ nbr,
                              int* __restrict__ cnt_arr,
                              float* __restrict__ pos_out,
                              long long* __restrict__ batch_out) {
    const int q    = blockIdx.x * 8 + (threadIdx.x >> 5);
    const int lane = threadIdx.x & 31;
    const int b    = q >> 10;
    const int ci   = idx[q];
    const float* pb = pos + (size_t)b * PP * 3;

    const float qx = pb[ci * 3 + 0];
    const float qy = pb[ci * 3 + 1];
    const float qz = pb[ci * 3 + 2];

    int count = 0;
    for (int base = 0; base < PP && count < KNEI; base += 32) {
        int i = base + lane;
        float dx = pb[i * 3 + 0] - qx;
        float dy = pb[i * 3 + 1] - qy;
        float dz = pb[i * 3 + 2] - qz;
        bool within = (dx * dx + dy * dy + dz * dz) <= RAD2;
        unsigned bal = __builtin_amdgcn_ballot_w32(within);
        int pre  = __popc(bal & ((1u << lane) - 1u));
        int slot = count + pre;
        if (within && slot < KNEI) nbr[q * KNEI + slot] = i;
        count += __popc(bal);
    }
    count = count < KNEI ? count : KNEI;
    if (lane == 0)  cnt_arr[q] = count;
    if (lane < 3)   pos_out[q * 3 + lane] = pb[ci * 3 + lane];
    if (lane == 0)  batch_out[q] = (long long)b;
}

// =====================================================================
// Kernel 3: grouped MLP + masked max-pool, bf16 WMMA, one WG / centroid.
// All weight buffers + neighbor features stream in via
// global_load_async_to_lds_b128; ASYNCcnt's in-order completion lets us
// overlap layer-1/2 WMMA with the W2/W3 transfers using partial waits.
//   per-wave async issue order:  W1 (6) | A-rows (<=2) | W2 (8) | W3 (16)
//   wait<=24> -> W1+A resident, wait<=16> -> W2 resident, wait<=0> -> W3.
// =====================================================================
template <int NOUT, int KSTEPS, bool FINAL>
__device__ __forceinline__ void gemm_layer(const __bf16* __restrict__ Ab, int sa,
                                           const __bf16* __restrict__ Wb,
                                           const float*  __restrict__ bias,
                                           __bf16* __restrict__ Hout,
                                           float*  __restrict__ omax,
                                           int cnt, int lane, int wave) {
    constexpr int NTW = NOUT / 32;
    const int mt  = wave & 3;
    const int nt0 = (wave >> 2) * NTW;
    const int m   = lane & 15;
    const int g   = lane >> 4;

    v8f acc[NTW] = {};

#pragma unroll
    for (int ks = 0; ks < KSTEPS; ++ks) {
        FragBF fa;
        const __bf16* arow = Ab + (mt * 16 + m) * sa + ks * 32;
        fa.q[0] = *(const uint4*)(arow + g * 8);
        fa.q[1] = *(const uint4*)(arow + 16 + g * 8);
#pragma unroll
        for (int tix = 0; tix < NTW; ++tix) {
            FragBF fb;
            const __bf16* bcol = Wb + (ks * 32) * NOUT + (nt0 + tix) * 16 + m;
#pragma unroll
            for (int s = 0; s < 16; ++s) {
                const int kk = (s < 8) ? (g * 8 + s) : (16 + g * 8 + (s - 8));
                fb.v[s] = bcol[kk * NOUT];
            }
            acc[tix] = __builtin_amdgcn_wmma_f32_16x16x32_bf16(
                false, fa.v, false, fb.v, (short)0, acc[tix], false, false);
        }
    }

#pragma unroll
    for (int tix = 0; tix < NTW; ++tix) {
        const int n  = (nt0 + tix) * 16 + m;
        const float bn = bias[n];
        if (!FINAL) {
#pragma unroll
            for (int r = 0; r < 8; ++r) {
                int mm = mt * 16 + g * 8 + r;
                float v = fmaxf(acc[tix][r] + bn, 0.0f);
                Hout[mm * NOUT + n] = f32_to_bf16(v);
            }
        } else {
            float lm = 0.0f;
#pragma unroll
            for (int r = 0; r < 8; ++r) {
                int mm = mt * 16 + g * 8 + r;
                if (mm < cnt) lm = fmaxf(lm, fmaxf(acc[tix][r] + bn, 0.0f));
            }
            atomicMax((int*)&omax[n], __float_as_int(lm));   // relu >= 0
        }
    }
}

__global__ void mlp_pool_kernel(const __bf16* __restrict__ xbf,
                                const float* __restrict__ pos,
                                const int* __restrict__ idx,
                                const int* __restrict__ nbr,
                                const int* __restrict__ cnt_arr,
                                const __bf16* __restrict__ wbf,
                                const float* __restrict__ b1,
                                const float* __restrict__ b2,
                                const float* __restrict__ b3,
                                float* __restrict__ xout) {
    extern __shared__ char smem[];
    __bf16* A   = (__bf16*)smem;               // [64][96]
    __bf16* Hb1 = A   + 64 * KPAD;             // [64][128]
    __bf16* Hb2 = Hb1 + 64 * H1DIM;            // [64][128]
    __bf16* W1l = Hb2 + 64 * H2DIM;            // [96][128]
    __bf16* W2l = W1l + W1ELEM;                // [128][128]
    __bf16* W3l = W2l + W2ELEM;                // [128][256]
    float*  b1l = (float*)(W3l + W3ELEM);      // [128]
    float*  b2l = b1l + H1DIM;                 // [128]
    float*  b3l = b2l + H2DIM;                 // [256]
    float*  omax = b3l + H3DIM;                // [256]

    const int q    = blockIdx.x;
    const int b    = q >> 10;
    const int t    = threadIdx.x;              // 0..255
    const int lane = t & 31;
    const int wave = t >> 5;

    const int cnt = cnt_arr[q];
    const int ci  = idx[q];
    const float* pb = pos + (size_t)b * PP * 3;

    // ---- issue W1 asyncs (6 instructions per wave, uniform) ----
    for (int i = t * 8; i < W1ELEM; i += 256 * 8)
        async_copy_b128(W1l + i, wbf + W1OFF + i);

    // ---- A matrix: async x-rows + ds zero/pos-diff tail (disjoint) ----
    {
        const int r   = t >> 2;                // neighbor row 0..63
        const int seg = t & 3;
        const int ni  = (r < cnt) ? nbr[q * KNEI + r] : -1;
        __bf16* arow = A + r * KPAD;

        // tail chunk: cols 64+8*seg .. 71+8*seg  (pos-diff in seg 0, zeros else)
        union { __bf16 h[8]; uint4 qv; } tail;
#pragma unroll
        for (int j = 0; j < 8; ++j) tail.h[j] = __builtin_bit_cast(__bf16, (unsigned short)0);
        if (seg == 0 && ni >= 0) {
            const float q0 = pb[ci * 3 + 0], q1 = pb[ci * 3 + 1], q2 = pb[ci * 3 + 2];
            tail.h[0] = f32_to_bf16(pb[ni * 3 + 0] - q0);
            tail.h[1] = f32_to_bf16(pb[ni * 3 + 1] - q1);
            tail.h[2] = f32_to_bf16(pb[ni * 3 + 2] - q2);
        }
        *(uint4*)(arow + CIN + 8 * seg) = tail.qv;

        if (ni >= 0) {
            const __bf16* src = xbf + (size_t)ni * CIN + seg * 16;
            async_copy_b128(arow + seg * 16,     src);
            async_copy_b128(arow + seg * 16 + 8, src + 8);
        } else {
            uint4 z = {0, 0, 0, 0};
            *(uint4*)(arow + seg * 16)     = z;
            *(uint4*)(arow + seg * 16 + 8) = z;
        }
    }

    // ---- issue W2 (8/wave) and W3 (16/wave) asyncs ----
    for (int i = t * 8; i < W2ELEM; i += 256 * 8)
        async_copy_b128(W2l + i, wbf + W2OFF + i);
    for (int i = t * 8; i < W3ELEM; i += 256 * 8)
        async_copy_b128(W3l + i, wbf + W3OFF + i);

    // biases + pool init (plain loads, overlap with async streams)
    if (t < H1DIM) b1l[t] = b1[t];
    if (t < H2DIM) b2l[t] = b2[t];
    b3l[t]  = b3[t];
    omax[t] = 0.0f;

    // ---- layer 1: [64x96] @ [96x128]  (W2/W3 still in flight) ----
    wait_asynccnt<24>();                       // W1 + A resident
    __syncthreads();
    gemm_layer<H1DIM, KPAD / 32, false>(A, KPAD, W1l, b1l, Hb1, nullptr, cnt, lane, wave);

    // ---- layer 2: [64x128] @ [128x128] (W3 still in flight) ----
    wait_asynccnt<16>();                       // W2 resident
    __syncthreads();
    gemm_layer<H2DIM, H1DIM / 32, false>(Hb1, H1DIM, W2l, b2l, Hb2, nullptr, cnt, lane, wave);

    // ---- layer 3: [64x128] @ [128x256] + masked max-pool ----
    wait_asynccnt<0>();                        // W3 resident
    __syncthreads();
    gemm_layer<H3DIM, H2DIM / 32, true>(Hb2, H2DIM, W3l, b3l, nullptr, omax, cnt, lane, wave);
    __syncthreads();

    xout[(size_t)q * H3DIM + t] = omax[t];
}

// =====================================================================
// host-side launcher
// =====================================================================
extern "C" void kernel_launch(void* const* d_in, const int* in_sizes, int n_in,
                              void* d_out, int out_size, void* d_ws, size_t ws_size,
                              hipStream_t stream) {
    const float* x   = (const float*)d_in[0];
    const float* pos = (const float*)d_in[1];
    // d_in[2] = batch (int64): deterministic, recomputed on device
    const float* W1 = (const float*)d_in[3];
    const float* b1 = (const float*)d_in[4];
    const float* W2 = (const float*)d_in[5];
    const float* b2 = (const float*)d_in[6];
    const float* W3 = (const float*)d_in[7];
    const float* b3 = (const float*)d_in[8];

    // workspace carve (16B-aligned sections)
    int*    idx = (int*)d_ws;                          // [NQ_TOT]
    int*    nbr = idx + NQ_TOT;                        // [NQ_TOT*KNEI]
    int*    cnt = nbr + NQ_TOT * KNEI;                 // [NQ_TOT]
    __bf16* xbf = (__bf16*)(cnt + NQ_TOT);             // [B*P*CIN]
    __bf16* wbf = xbf + (size_t)BB * PP * CIN;         // [WELEM]

    // output carve: x_out f32 [NQ*256] | pos_out f32 [NQ*3] | batch int64 [NQ]
    float*     outf      = (float*)d_out;
    float*     pos_out   = outf + (size_t)NQ_TOT * H3DIM;
    long long* batch_out = (long long*)(pos_out + (size_t)NQ_TOT * 3);

    // 0) one-time bf16 conversion of x and padded weights
    convert_bf16_kernel<<<2048, 256, 0, stream>>>(x, W1, W2, W3, xbf, wbf);

    // 1) FPS: one WG per cloud
    size_t fps_lds = (size_t)3 * PP * sizeof(float) + 32 * sizeof(unsigned long long);
    fps_kernel<<<BB, 1024, fps_lds, stream>>>(pos, idx);

    // 2) radius search: one wave per query
    radius_kernel<<<NQ_TOT / 8, 256, 0, stream>>>(pos, idx, nbr, cnt, pos_out, batch_out);

    // 3) MLP + max-pool: one WG per centroid, bf16 WMMA + async LDS staging
    size_t mlp_lds = (size_t)(64 * KPAD + 64 * H1DIM + 64 * H2DIM + WELEM) * sizeof(__bf16)
                   + (size_t)(H1DIM + H2DIM + H3DIM + H3DIM) * sizeof(float);
    mlp_pool_kernel<<<NQ_TOT, 256, mlp_lds, stream>>>(xbf, pos, idx, nbr, cnt,
                                                      wbf, b1, b2, b3, outf);
}
